// Attention_87205015978193
// MI455X (gfx1250) — matrix operand order repaired and checked by reading, not compiled, not run on hardware
//
#include <hip/hip_runtime.h>
#include <hip/hip_fp16.h>

typedef __attribute__((ext_vector_type(16))) _Float16 v16h;
typedef __attribute__((ext_vector_type(8)))  _Float16 v8h;
typedef __attribute__((ext_vector_type(8)))  float    v8f;

#define BB   4
#define SS   2048
#define DDIM 1024
#define HH   16
#define DKK  64
#define NROW (BB*SS)   // 8192

// ---------------------------------------------------------------------------
// WMMA fragment loaders (layouts per CDNA5 ISA 7.12.2, wave32)
// A 16x32 f16: lane L holds row M=(L&15); halves 0..7 -> K = k0+(L>=16?8:0)+0..7
//              halves 8..15 -> K = k0+(L>=16?8:0)+16..23
// B 32x16 f16 (source stored [N][K] row-major): lane L holds row N=(L&15);
//              halves 0..7 -> K = k0+(L>=16?8:0)+0..7, 8..15 -> +16..23
// C/D 16x16 f32: lane L holds N=(L&15); VGPR r holds M = r + (L>=16?8:0)
// ---------------------------------------------------------------------------
__device__ __forceinline__ v16h frag_a(const _Float16* p, int ld, int lane) {
  const _Float16* q = p + (size_t)(lane & 15) * ld + ((lane & 16) ? 8 : 0);
  union { v16h v; v8h h[2]; } u;
  u.h[0] = *(const v8h*)(q);
  u.h[1] = *(const v8h*)(q + 16);
  return u.v;
}

__device__ __forceinline__ v16h frag_b(const _Float16* p, int ld, int lane) {
  const _Float16* q = p + (size_t)(lane & 15) * ld + ((lane & 16) ? 8 : 0);
  union { v16h v; v8h h[2]; } u;
  u.h[0] = *(const v8h*)(q);
  u.h[1] = *(const v8h*)(q + 16);
  return u.v;
}

__device__ __forceinline__ v8f wmma32(v16h a, v16h b, v8f c) {
  return __builtin_amdgcn_wmma_f32_16x16x32_f16(false, a, false, b, (short)0, c,
                                                false, false);
}

// ---------------------------------------------------------------------------
// Kernel 1a: fp32 -> f16 convert
// ---------------------------------------------------------------------------
__global__ void k_cvt(const float* __restrict__ x, _Float16* __restrict__ y, int n) {
  int i = blockIdx.x * 256 + threadIdx.x;
  if (i < n) y[i] = (_Float16)x[i];
}

// Kernel 1b: W [K=1024][N=1024] fp32 -> WT [N][K] f16
__global__ void k_wt(const float* __restrict__ w, _Float16* __restrict__ wt) {
  int i = blockIdx.x * 256 + threadIdx.x;     // 1M elements
  int n = i >> 10, k = i & 1023;
  wt[(size_t)n * 1024 + k] = (_Float16)w[(size_t)k * 1024 + n];
}

// ---------------------------------------------------------------------------
// Software-pipelined 32x64 GEMM core: D[2][4] accumulators, register double
// buffering of next k-step fragments so global_load_b128 overlaps v_wmma.
// A: rows [m0, m0+32), row-major ldA halves.  B: rows [n0, n0+64) of [N][K].
// ---------------------------------------------------------------------------
__device__ __forceinline__ void gemm32x64(const _Float16* A, const _Float16* B,
                                          int lane, v8f acc[2][4]) {
  v16h a0 = frag_a(A, DDIM, lane);
  v16h a1 = frag_a(A + (size_t)16 * DDIM, DDIM, lane);
  v16h b0 = frag_b(B, DDIM, lane);
  v16h b1 = frag_b(B + (size_t)16 * DDIM, DDIM, lane);
  v16h b2 = frag_b(B + (size_t)32 * DDIM, DDIM, lane);
  v16h b3 = frag_b(B + (size_t)48 * DDIM, DDIM, lane);

  for (int k = 0; k < DDIM; k += 32) {
    const int kn = (k + 32) & (DDIM - 1);   // wraps to 0 on last iter (dead)
    v16h na0 = frag_a(A + kn, DDIM, lane);
    v16h na1 = frag_a(A + (size_t)16 * DDIM + kn, DDIM, lane);
    v16h nb0 = frag_b(B + kn, DDIM, lane);
    v16h nb1 = frag_b(B + (size_t)16 * DDIM + kn, DDIM, lane);
    v16h nb2 = frag_b(B + (size_t)32 * DDIM + kn, DDIM, lane);
    v16h nb3 = frag_b(B + (size_t)48 * DDIM + kn, DDIM, lane);

    acc[0][0] = wmma32(a0, b0, acc[0][0]);
    acc[0][1] = wmma32(a0, b1, acc[0][1]);
    acc[0][2] = wmma32(a0, b2, acc[0][2]);
    acc[0][3] = wmma32(a0, b3, acc[0][3]);
    acc[1][0] = wmma32(a1, b0, acc[1][0]);
    acc[1][1] = wmma32(a1, b1, acc[1][1]);
    acc[1][2] = wmma32(a1, b2, acc[1][2]);
    acc[1][3] = wmma32(a1, b3, acc[1][3]);

    a0 = na0; a1 = na1; b0 = nb0; b1 = nb1; b2 = nb2; b3 = nb3;
  }
}

// ---------------------------------------------------------------------------
// Kernel 2: fused QKV projection.  Wave = 32(M) x 64(N) tile.
// grid.x: M blocks of 128 (64), grid.y: N blocks of 64 over 3*1024 (48)
// Q,K -> [b,h,s,dk] f16 ; V -> [b,h,dk,s] f16 (pre-transposed for P@V)
// ---------------------------------------------------------------------------
__global__ void k_qkv(const _Float16* __restrict__ Xh,
                      const _Float16* __restrict__ WqT,
                      const _Float16* __restrict__ WkT,
                      const _Float16* __restrict__ WvT,
                      const float* __restrict__ bq, const float* __restrict__ bk,
                      const float* __restrict__ bv,
                      _Float16* __restrict__ Qh, _Float16* __restrict__ Kh,
                      _Float16* __restrict__ Vt) {
  const int lane = threadIdx.x & 31;
  const int wave = threadIdx.x >> 5;
  const int m0   = blockIdx.x * 128 + wave * 32;
  const int nblk = blockIdx.y * 64;            // 0..3071
  const int mode = nblk >> 10;                 // 0=Q 1=K 2=V
  const int n0   = nblk & 1023;
  const _Float16* WT  = (mode == 0) ? WqT : (mode == 1) ? WkT : WvT;
  const float*   bias = (mode == 0) ? bq  : (mode == 1) ? bk  : bv;

  v8f acc[2][4] = {};
  gemm32x64(Xh + (size_t)m0 * DDIM, WT + (size_t)n0 * DDIM, lane, acc);

  const int hi = (lane & 16) ? 8 : 0;
#pragma unroll
  for (int t = 0; t < 2; ++t) {
#pragma unroll
    for (int nt = 0; nt < 4; ++nt) {
      int n = n0 + nt * 16 + (lane & 15);
      float bsv = bias[n];
      int h = n >> 6, dk = n & 63;
#pragma unroll
      for (int r = 0; r < 8; ++r) {
        int m = m0 + t * 16 + r + hi;
        int b_ = m >> 11, s = m & (SS - 1);
        _Float16 val = (_Float16)(acc[t][nt][r] + bsv);
        if (mode == 2)
          Vt[(((size_t)b_ * HH + h) * DKK + dk) * SS + s] = val;
        else if (mode == 0)
          Qh[(((size_t)b_ * HH + h) * SS + s) * DKK + dk] = val;
        else
          Kh[(((size_t)b_ * HH + h) * SS + s) * DKK + dk] = val;
      }
    }
  }
}

// ---------------------------------------------------------------------------
// Kernel 3: flash attention.  Wave = 16 queries; 64-key blocks; online softmax.
// grid.x = S/64 (32), grid.y = B*H (64), block = 128 threads (4 waves)
// V fragments loaded before the softmax VALU block (latency hiding);
// next key-block K/V warmed with global_prefetch_b8.
// ---------------------------------------------------------------------------
__global__ void k_attn(const _Float16* __restrict__ Qh,
                       const _Float16* __restrict__ Kh,
                       const _Float16* __restrict__ Vt,
                       const float* __restrict__ mask,
                       _Float16* __restrict__ Ctx) {
  __shared__ _Float16 Pt[4][16 * 64];
  const int lane = threadIdx.x & 31;
  const int wave = threadIdx.x >> 5;
  const int q0   = blockIdx.x * 64 + wave * 16;
  const int bh   = blockIdx.y;
  const int b    = bh >> 4;
  const int h    = bh & 15;
  const float scale = 0.125f;  // 1/sqrt(64)

  const _Float16* Qp = Qh + (size_t)bh * SS * DKK;
  const _Float16* Kp = Kh + (size_t)bh * SS * DKK;
  const _Float16* Vp = Vt + (size_t)bh * DKK * SS;
  const float*    mk = mask + (size_t)b * SS;

  v16h qa0 = frag_a(Qp + (size_t)q0 * DKK + 0,  DKK, lane);
  v16h qa1 = frag_a(Qp + (size_t)q0 * DKK + 32, DKK, lane);

  v8f ctx[4] = {};
  float mrow[8], lrow[8];
#pragma unroll
  for (int r = 0; r < 8; ++r) { mrow[r] = -1e30f; lrow[r] = 0.f; }

  _Float16* P = &Pt[wave][0];
  const int hi = (lane & 16) ? 8 : 0;

  for (int kb = 0; kb < SS; kb += 64) {
    // ---- scores: S = Q K^T  (16 x 64 tile); all K loads before WMMAs ----
    v16h kf0[4], kf1[4];
#pragma unroll
    for (int nt = 0; nt < 4; ++nt) {
      kf0[nt] = frag_b(Kp + (size_t)(kb + nt * 16) * DKK + 0,  DKK, lane);
      kf1[nt] = frag_b(Kp + (size_t)(kb + nt * 16) * DKK + 32, DKK, lane);
    }
    v8f sc[4] = {};
#pragma unroll
    for (int nt = 0; nt < 4; ++nt) {
      sc[nt] = wmma32(qa0, kf0[nt], sc[nt]);
      sc[nt] = wmma32(qa1, kf1[nt], sc[nt]);
    }

    // warm next key block while this block's softmax runs
    if (kb + 64 < SS) {
      __builtin_prefetch(Kp + (size_t)(kb + 64 + lane) * DKK, 0, 1);
      __builtin_prefetch(Kp + (size_t)(kb + 96 + lane) * DKK, 0, 1);
      __builtin_prefetch(Vp + (size_t)lane * SS + kb + 64, 0, 1);
      __builtin_prefetch(Vp + (size_t)(lane + 32) * SS + kb + 64, 0, 1);
    }

    // ---- V fragments issued early: global loads overlap softmax VALU ----
    v16h vf0[4], vf1[4];
#pragma unroll
    for (int nt = 0; nt < 4; ++nt) {
      vf0[nt] = frag_b(Vp + (size_t)(nt * 16) * SS + kb + 0,  SS, lane);
      vf1[nt] = frag_b(Vp + (size_t)(nt * 16) * SS + kb + 32, SS, lane);
    }

    float mv[4];
#pragma unroll
    for (int nt = 0; nt < 4; ++nt) mv[nt] = mk[kb + nt * 16 + (lane & 15)];

    // ---- online softmax ----
    float newm[8];
#pragma unroll
    for (int r = 0; r < 8; ++r) {
      float mx = mrow[r];
#pragma unroll
      for (int nt = 0; nt < 4; ++nt) {
        float s = sc[nt][r] * scale + mv[nt];
        sc[nt][r] = s;
        mx = fmaxf(mx, s);
      }
      mx = fmaxf(mx, __shfl_xor(mx, 1, 32));
      mx = fmaxf(mx, __shfl_xor(mx, 2, 32));
      mx = fmaxf(mx, __shfl_xor(mx, 4, 32));
      mx = fmaxf(mx, __shfl_xor(mx, 8, 32));
      newm[r] = mx;
    }
#pragma unroll
    for (int r = 0; r < 8; ++r) {
      float alpha = __expf(mrow[r] - newm[r]);
      float rs = 0.f;
#pragma unroll
      for (int nt = 0; nt < 4; ++nt) {
        float p = __expf(sc[nt][r] - newm[r]);
        rs += p;
        P[(r + hi) * 64 + nt * 16 + (lane & 15)] = (_Float16)p;  // C->A reshape
      }
      rs += __shfl_xor(rs, 1, 32);
      rs += __shfl_xor(rs, 2, 32);
      rs += __shfl_xor(rs, 4, 32);
      rs += __shfl_xor(rs, 8, 32);
      lrow[r] = lrow[r] * alpha + rs;
      mrow[r] = newm[r];
#pragma unroll
      for (int nt = 0; nt < 4; ++nt) ctx[nt][r] *= alpha;
    }

    // same-wave LDS RAW: DS ops are in-order per wave; barrier the compiler too
    asm volatile("s_wait_dscnt 0" ::: "memory");

    // ---- ctx += P V  (contraction over 64 keys) ----
    v16h pa0 = frag_a(P + 0,  64, lane);
    v16h pa1 = frag_a(P + 32, 64, lane);
#pragma unroll
    for (int nt = 0; nt < 4; ++nt) {
      ctx[nt] = wmma32(pa0, vf0[nt], ctx[nt]);
      ctx[nt] = wmma32(pa1, vf1[nt], ctx[nt]);
    }
    asm volatile("" ::: "memory");  // keep this block's LDS reads before next stores
  }

  // ---- finalize: divide by row sum, store ctx f16 [b, s, h*64+dk] ----
#pragma unroll
  for (int r = 0; r < 8; ++r) {
    float inv = 1.f / lrow[r];
    int q = q0 + r + hi;
#pragma unroll
    for (int nt = 0; nt < 4; ++nt) {
      int dk = nt * 16 + (lane & 15);
      Ctx[((size_t)b * SS + q) * DDIM + h * DKK + dk] = (_Float16)(ctx[nt][r] * inv);
    }
  }
}

// ---------------------------------------------------------------------------
// Kernel 4: output projection, fp32 result + bias.  Wave = 32x64 tile.
// ---------------------------------------------------------------------------
__global__ void k_out(const _Float16* __restrict__ Ctxh,
                      const _Float16* __restrict__ WoT,
                      const float* __restrict__ bo, float* __restrict__ out) {
  const int lane = threadIdx.x & 31;
  const int wave = threadIdx.x >> 5;
  const int m0 = blockIdx.x * 128 + wave * 32;
  const int n0 = blockIdx.y * 64;

  v8f acc[2][4] = {};
  gemm32x64(Ctxh + (size_t)m0 * DDIM, WoT + (size_t)n0 * DDIM, lane, acc);

  const int hi = (lane & 16) ? 8 : 0;
#pragma unroll
  for (int t = 0; t < 2; ++t) {
#pragma unroll
    for (int nt = 0; nt < 4; ++nt) {
      int n = n0 + nt * 16 + (lane & 15);
      float bb = bo[n];
#pragma unroll
      for (int r = 0; r < 8; ++r)
        out[(size_t)(m0 + t * 16 + r + hi) * DDIM + n] = acc[t][nt][r] + bb;
    }
  }
}

// ---------------------------------------------------------------------------
extern "C" void kernel_launch(void* const* d_in, const int* in_sizes, int n_in,
                              void* d_out, int out_size, void* d_ws, size_t ws_size,
                              hipStream_t stream) {
  const float* hs   = (const float*)d_in[0];  // [B,S,D]
  const float* mask = (const float*)d_in[1];  // [B,1,1,S]
  const float* Wq   = (const float*)d_in[2];
  const float* bq   = (const float*)d_in[3];
  const float* Wk   = (const float*)d_in[4];
  const float* bk   = (const float*)d_in[5];
  const float* Wv   = (const float*)d_in[6];
  const float* bv   = (const float*)d_in[7];
  const float* Wo   = (const float*)d_in[8];
  const float* bo   = (const float*)d_in[9];
  float* out = (float*)d_out;

  char* ws = (char*)d_ws;
  const size_t szX  = (size_t)NROW * DDIM * 2;        // 16 MB
  const size_t szW  = (size_t)DDIM * DDIM * 2;        // 2 MB
  const size_t szQ  = (size_t)BB * HH * SS * DKK * 2; // 16 MB
  _Float16* Xh   = (_Float16*)(ws);
  _Float16* WqT  = (_Float16*)(ws + szX);
  _Float16* WkT  = (_Float16*)(ws + szX + szW);
  _Float16* WvT  = (_Float16*)(ws + szX + 2 * szW);
  _Float16* WoT  = (_Float16*)(ws + szX + 3 * szW);
  _Float16* Qh   = (_Float16*)(ws + szX + 4 * szW);
  _Float16* Kh   = (_Float16*)(ws + szX + 4 * szW + szQ);
  _Float16* Vt   = (_Float16*)(ws + szX + 4 * szW + 2 * szQ);
  _Float16* Ctxh = (_Float16*)(ws + szX + 4 * szW + 3 * szQ);

  // 1) convert X, transpose+convert weights
  {
    int n = NROW * DDIM;
    k_cvt<<<dim3((n + 255) / 256), dim3(256), 0, stream>>>(hs, Xh, n);
    dim3 g(DDIM * DDIM / 256);
    k_wt<<<g, dim3(256), 0, stream>>>(Wq, WqT);
    k_wt<<<g, dim3(256), 0, stream>>>(Wk, WkT);
    k_wt<<<g, dim3(256), 0, stream>>>(Wv, WvT);
    k_wt<<<g, dim3(256), 0, stream>>>(Wo, WoT);
  }
  // 2) fused QKV projection
  k_qkv<<<dim3(NROW / 128, 3 * DDIM / 64), dim3(128), 0, stream>>>(
      Xh, WqT, WkT, WvT, bq, bk, bv, Qh, Kh, Vt);
  // 3) flash attention
  k_attn<<<dim3(SS / 64, BB * HH), dim3(128), 0, stream>>>(Qh, Kh, Vt, mask, Ctxh);
  // 4) output projection
  k_out<<<dim3(NROW / 128, DDIM / 64), dim3(128), 0, stream>>>(Ctxh, WoT, bo, out);
}
